// CharacterEncoder_7172595384970
// MI455X (gfx1250) — matrix operand order repaired and checked by reading, not compiled, not run on hardware
//
#include <hip/hip_runtime.h>
#include <hip/hip_bf16.h>

#define S 2048
#define D 2048
#define H 16
#define HD 128
#define LN_EPS 1e-5f

typedef __attribute__((ext_vector_type(16))) __bf16 v16bf;
typedef __attribute__((ext_vector_type(8)))  __bf16 v8bf;
typedef __attribute__((ext_vector_type(8)))  float  v8f;

__device__ __forceinline__ unsigned short f2bf(float f) {
    union { float f; unsigned int u; } v; v.f = f;
    unsigned int u = v.u;
    u += 0x7fffu + ((u >> 16) & 1u);   // round-to-nearest-even
    return (unsigned short)(u >> 16);
}

__device__ __forceinline__ v8f v8f_zero() {
    v8f z;
#pragma unroll
    for (int i = 0; i < 8; i++) z[i] = 0.0f;
    return z;
}

__device__ __forceinline__ v16bf combine16(v8bf lo, v8bf hi) {
    v16bf a;
#pragma unroll
    for (int i = 0; i < 8; i++) { a[i] = lo[i]; a[i + 8] = hi[i]; }
    return a;
}

__device__ __forceinline__ v8f wmma_bf16(v16bf a, v16bf b, v8f c) {
    return __builtin_amdgcn_wmma_f32_16x16x32_bf16(
        /*neg_a=*/false, a, /*neg_b=*/false, b,
        /*c_mod=*/(short)0, c, /*reuse_a=*/false, /*reuse_b=*/false);
}

// Issue one 16-byte async global->LDS copy (CDNA5 TDM-lite path, ASYNCcnt).
__device__ __forceinline__ void async_copy_b128(const unsigned short* gsrc,
                                                unsigned short* ldst) {
    unsigned lds_off = (unsigned)(uintptr_t)ldst;            // low 32 bits = LDS offset
    unsigned long long ga = (unsigned long long)(uintptr_t)gsrc;
    asm volatile("global_load_async_to_lds_b128 %0, %1, off"
                 :: "v"(lds_off), "v"(ga) : "memory");
}

__device__ __forceinline__ void wait_asynccnt0() {
    asm volatile("s_wait_asynccnt 0x0" ::: "memory");
}

// ---------------------------------------------------------------------------
// Kernel 0: convert in_w (3D x D) and out_w (D x D) fp32 -> bf16
// ---------------------------------------------------------------------------
__global__ __launch_bounds__(256) void convert_w_kernel(
    const float* __restrict__ in_w, const float* __restrict__ out_w,
    unsigned short* __restrict__ in_w_bf, unsigned short* __restrict__ out_w_bf)
{
    size_t idx = (size_t)blockIdx.x * 256 + threadIdx.x;
    const size_t n1 = (size_t)3 * D * D;
    const size_t n2 = (size_t)D * D;
    if (idx < n1) in_w_bf[idx] = f2bf(in_w[idx]);
    else {
        size_t j = idx - n1;
        if (j < n2) out_w_bf[j] = f2bf(out_w[j]);
    }
}

// ---------------------------------------------------------------------------
// Kernel 1: fused embedding add + LayerNorm.  One block (256 thr) per row.
// ---------------------------------------------------------------------------
__global__ __launch_bounds__(256) void embed_ln_kernel(
    const float* __restrict__ byte_repr, const int* __restrict__ categories,
    const int* __restrict__ casev, const int* __restrict__ combining,
    const float* __restrict__ cat_emb, const float* __restrict__ case_emb,
    const float* __restrict__ comb_emb,
    const float* __restrict__ ln_g, const float* __restrict__ ln_b,
    float* __restrict__ xn_f32, unsigned short* __restrict__ xn_bf)
{
    const int s = blockIdx.x;
    const int tid = threadIdx.x;
    const float* br = byte_repr + (size_t)s * D;
    const float* ce = cat_emb  + (size_t)categories[s] * D;
    const float* se = case_emb + (size_t)casev[s]      * D;
    const float* be = comb_emb + (size_t)combining[s]  * D;

    float x[8];
    float sum = 0.0f, sumsq = 0.0f;
#pragma unroll
    for (int k = 0; k < 8; k++) {
        int i = tid + k * 256;
        float v = br[i] + ce[i] + se[i] + be[i];
        x[k] = v; sum += v; sumsq += v * v;
    }
    __shared__ float red[256], red2[256];
    red[tid] = sum; red2[tid] = sumsq;
    __syncthreads();
    for (int off = 128; off > 0; off >>= 1) {
        if (tid < off) { red[tid] += red[tid + off]; red2[tid] += red2[tid + off]; }
        __syncthreads();
    }
    float mu  = red[0] * (1.0f / D);
    float var = red2[0] * (1.0f / D) - mu * mu;
    float inv = rsqrtf(var + LN_EPS);
#pragma unroll
    for (int k = 0; k < 8; k++) {
        int i = tid + k * 256;
        float xn = (x[k] - mu) * inv * ln_g[i] + ln_b[i];
        xn_f32[(size_t)s * D + i] = xn;
        xn_bf [(size_t)s * D + i] = f2bf(xn);
    }
}

// ---------------------------------------------------------------------------
// Shared GEMM core: block = 8 waves, block tile = 256 rows x 64 cols.
// Each wave owns 32 rows x 64 cols (8 WMMAs / K-step).
// B tile (64 cols x 32 K, bf16 = 4KB) is double-buffered in LDS and filled
// with async global->LDS b128 copies (one 16B chunk per thread), overlapped
// with the current K-step's WMMAs.
//   LDS layout: Bsh[buf][col*32 + k]  (ushort)
//   stage chunk for thread i: col = i>>2, k-chunk = (i&3)*8  -> offset i*8
// ---------------------------------------------------------------------------
__device__ __forceinline__ void gemm_core_32x64(
    const unsigned short* __restrict__ A,   // [rows x D] bf16
    const unsigned short* __restrict__ Wb,  // [N x D]    bf16
    int row0, int col0, unsigned short* Bsh /* [2][2048] */,
    v8f acc[2][4])
{
    const int tid   = threadIdx.x;
    const int lane  = tid & 31;
    const int lhalf = lane >> 4, lmod = lane & 15;
    const int scol   = col0 + (tid >> 2);        // staging column for this thread
    const int schunk = (tid & 3) * 8;            // staging k-chunk (8 bf16 = 16B)

#pragma unroll
    for (int rb = 0; rb < 2; rb++)
#pragma unroll
        for (int t = 0; t < 4; t++) acc[rb][t] = v8f_zero();

    // prologue: stage k-step 0 into buffer 0
    async_copy_b128(Wb + (size_t)scol * D + schunk, Bsh + tid * 8);

    for (int kk = 0; kk < D / 32; kk++) {
        const int k0 = kk * 32;
        wait_asynccnt0();          // our 32 staged chunks for step kk landed
        __syncthreads();           // everyone's chunks visible; prev buffer free

        if (kk + 1 < D / 32) {     // stage next step into the other buffer
            const int nk0 = (kk + 1) * 32;
            unsigned short* nbuf = Bsh + ((kk + 1) & 1) * 2048;
            async_copy_b128(Wb + (size_t)scol * D + nk0 + schunk, nbuf + tid * 8);
        }

        // A fragments for this wave's two 16-row blocks
        const unsigned short* ar0 = A + (size_t)(row0 + lmod) * D + k0 + lhalf * 16;
        const unsigned short* ar1 = A + (size_t)(row0 + 16 + lmod) * D + k0 + lhalf * 16;
        v16bf a0 = combine16(*(const v8bf*)ar0, *(const v8bf*)(ar0 + 8));
        v16bf a1 = combine16(*(const v8bf*)ar1, *(const v8bf*)(ar1 + 8));

        const unsigned short* bb = Bsh + (kk & 1) * 2048;
#pragma unroll
        for (int t = 0; t < 4; t++) {
            const unsigned short* bp = bb + (t * 16 + lmod) * 32 + lhalf * 16;
            v16bf b = *(const v16bf*)bp;
            acc[0][t] = wmma_bf16(a0, b, acc[0][t]);
            acc[1][t] = wmma_bf16(a1, b, acc[1][t]);
        }
    }
}

// ---------------------------------------------------------------------------
// Kernel 2: QKV GEMM  qkv[s,n] = xn[s,:] . in_w[n,:] + in_b[n]
// Epilogue routes to q (pre-scaled by 1/sqrt(HD)), k, and v-transposed.
// grid: (S/256) * (3D/64) = 8 * 96 = 768 blocks
// ---------------------------------------------------------------------------
__global__ __launch_bounds__(256) void qkv_gemm_kernel(
    const unsigned short* __restrict__ xn_bf, const unsigned short* __restrict__ w_bf,
    const float* __restrict__ in_b,
    unsigned short* __restrict__ q_bf, unsigned short* __restrict__ k_bf,
    unsigned short* __restrict__ vT_bf)
{
    __shared__ __align__(32) unsigned short Bsh[2][64 * 32];
    const int NT = (3 * D) / 64;               // 96
    const int bc = blockIdx.x % NT;
    const int br = blockIdx.x / NT;            // 0..7
    const int wv = threadIdx.x >> 5;
    const int lane = threadIdx.x & 31;
    const int lhalf = lane >> 4, lmod = lane & 15;
    const int row0 = br * 256 + wv * 32;
    const int col0 = bc * 64;

    v8f acc[2][4];
    gemm_core_32x64(xn_bf, w_bf, row0, col0, &Bsh[0][0], acc);

    const float QSCALE = 0.08838834764831845f;   // 1/sqrt(128)
#pragma unroll
    for (int rb = 0; rb < 2; rb++)
#pragma unroll
        for (int t = 0; t < 4; t++) {
            const int n = col0 + t * 16 + lmod;
            const float bias = in_b[n];
#pragma unroll
            for (int r = 0; r < 8; r++) {
                const int srow = row0 + rb * 16 + lhalf * 8 + r;
                const float v = acc[rb][t][r] + bias;
                if (n < D)          q_bf [(size_t)srow * D + n]           = f2bf(v * QSCALE);
                else if (n < 2 * D) k_bf [(size_t)srow * D + (n - D)]     = f2bf(v);
                else                vT_bf[(size_t)(n - 2 * D) * S + srow] = f2bf(v);
            }
        }
}

// ---------------------------------------------------------------------------
// Kernel 3: flash-style attention. One wave per (head, 16-query tile).
// scores = q.k^T (q pre-scaled) + segmask(+1/+0); online softmax; o = P.V.
// ---------------------------------------------------------------------------
__global__ __launch_bounds__(256) void attn_kernel(
    const unsigned short* __restrict__ q_bf, const unsigned short* __restrict__ k_bf,
    const unsigned short* __restrict__ vT_bf, const int* __restrict__ seg_ids,
    unsigned short* __restrict__ o_bf)
{
    __shared__ __align__(32) unsigned short pst[8][16 * 32];
    const int wv   = threadIdx.x >> 5;
    const int lane = threadIdx.x & 31;
    const int task = (int)(blockIdx.x * 8 + wv);   // 0..2047
    const int h  = task >> 7;                      // /(S/16)
    const int qt = task & 127;
    const int lhalf = lane >> 4, lmod = lane & 15;

    v16bf qA[4];
    const unsigned short* qbase = q_bf + (size_t)(qt * 16 + lmod) * D + h * HD;
#pragma unroll
    for (int kk = 0; kk < 4; kk++) {
        const unsigned short* p = qbase + kk * 32 + lhalf * 16;
        qA[kk] = combine16(*(const v8bf*)p, *(const v8bf*)(p + 8));
    }
    int segq[8];
#pragma unroll
    for (int r = 0; r < 8; r++) segq[r] = seg_ids[qt * 16 + lhalf * 8 + r];

    float mrow[8], lrow[8];
#pragma unroll
    for (int r = 0; r < 8; r++) { mrow[r] = -3.0e38f; lrow[r] = 0.0f; }
    v8f oacc[8];
#pragma unroll
    for (int dt = 0; dt < 8; dt++) oacc[dt] = v8f_zero();

    unsigned short* myp = pst[wv];

    for (int jt2 = 0; jt2 < S / 32; jt2++) {
        v8f sc[2];
#pragma unroll
        for (int sub = 0; sub < 2; sub++) {
            sc[sub] = v8f_zero();
            const int j0 = jt2 * 32 + sub * 16;
            const unsigned short* kb = k_bf + (size_t)(j0 + lmod) * D + h * HD + lhalf * 16;
#pragma unroll
            for (int kk = 0; kk < 4; kk++) {
                v16bf b = *(const v16bf*)(kb + kk * 32);
                sc[sub] = wmma_bf16(qA[kk], b, sc[sub]);
            }
            const int segk = seg_ids[j0 + lmod];
#pragma unroll
            for (int r = 0; r < 8; r++)
                sc[sub][r] += (segq[r] == segk) ? 1.0f : 0.0f;
        }

        float corr[8];
#pragma unroll
        for (int r = 0; r < 8; r++) {
            float t = fmaxf(sc[0][r], sc[1][r]);
            t = fmaxf(t, __shfl_xor(t, 1, 32));
            t = fmaxf(t, __shfl_xor(t, 2, 32));
            t = fmaxf(t, __shfl_xor(t, 4, 32));
            t = fmaxf(t, __shfl_xor(t, 8, 32));
            const float mnew = fmaxf(mrow[r], t);
            const float c  = __expf(mrow[r] - mnew);
            const float p0 = __expf(sc[0][r] - mnew);
            const float p1 = __expf(sc[1][r] - mnew);
            sc[0][r] = p0; sc[1][r] = p1;
            float ps = p0 + p1;
            ps += __shfl_xor(ps, 1, 32);
            ps += __shfl_xor(ps, 2, 32);
            ps += __shfl_xor(ps, 4, 32);
            ps += __shfl_xor(ps, 8, 32);
            lrow[r] = lrow[r] * c + ps;
            mrow[r] = mnew; corr[r] = c;
        }
#pragma unroll
        for (int dt = 0; dt < 8; dt++)
#pragma unroll
            for (int r = 0; r < 8; r++) oacc[dt][r] *= corr[r];

        // D-layout -> A-layout transpose of P through per-wave LDS
#pragma unroll
        for (int sub = 0; sub < 2; sub++)
#pragma unroll
            for (int r = 0; r < 8; r++)
                myp[(lhalf * 8 + r) * 32 + sub * 16 + lmod] = f2bf(sc[sub][r]);

        asm volatile("s_wait_dscnt 0x0" ::: "memory");

        const unsigned short* pr = myp + lmod * 32 + lhalf * 16;
        v16bf pa = combine16(*(const v8bf*)pr, *(const v8bf*)(pr + 8));

        const unsigned short* vb = vT_bf + (size_t)(h * HD + lmod) * S + jt2 * 32 + lhalf * 16;
#pragma unroll
        for (int dt = 0; dt < 8; dt++) {
            v16bf b = *(const v16bf*)(vb + (size_t)dt * 16 * S);
            oacc[dt] = wmma_bf16(pa, b, oacc[dt]);
        }
    }

#pragma unroll
    for (int dt = 0; dt < 8; dt++)
#pragma unroll
        for (int r = 0; r < 8; r++) {
            const int srow = qt * 16 + lhalf * 8 + r;
            const float v = oacc[dt][r] / lrow[r];
            o_bf[(size_t)srow * D + h * HD + dt * 16 + lmod] = f2bf(v);
        }
}

// ---------------------------------------------------------------------------
// Kernel 4: output projection + bias + residual.
// grid: (S/256) * (D/64) = 8 * 32 = 256 blocks
// ---------------------------------------------------------------------------
__global__ __launch_bounds__(256) void out_gemm_kernel(
    const unsigned short* __restrict__ o_bf, const unsigned short* __restrict__ w_bf,
    const float* __restrict__ out_b, const float* __restrict__ xn_f32,
    float* __restrict__ out)
{
    __shared__ __align__(32) unsigned short Bsh[2][64 * 32];
    const int NT = D / 64;                 // 32
    const int bc = blockIdx.x % NT;
    const int br = blockIdx.x / NT;        // 0..7
    const int wv = threadIdx.x >> 5;
    const int lane = threadIdx.x & 31;
    const int lhalf = lane >> 4, lmod = lane & 15;
    const int row0 = br * 256 + wv * 32;
    const int col0 = bc * 64;

    v8f acc[2][4];
    gemm_core_32x64(o_bf, w_bf, row0, col0, &Bsh[0][0], acc);

#pragma unroll
    for (int rb = 0; rb < 2; rb++)
#pragma unroll
        for (int t = 0; t < 4; t++) {
            const int n = col0 + t * 16 + lmod;
            const float bias = out_b[n];
#pragma unroll
            for (int r = 0; r < 8; r++) {
                const int srow = row0 + rb * 16 + lhalf * 8 + r;
                out[(size_t)srow * D + n] =
                    acc[rb][t][r] + bias + xn_f32[(size_t)srow * D + n];
            }
        }
}

// ---------------------------------------------------------------------------
extern "C" void kernel_launch(void* const* d_in, const int* in_sizes, int n_in,
                              void* d_out, int out_size, void* d_ws, size_t ws_size,
                              hipStream_t stream) {
    (void)in_sizes; (void)n_in; (void)out_size; (void)ws_size;
    const float* byte_repr = (const float*)d_in[0];
    const int*   categories = (const int*)d_in[1];
    const int*   casev      = (const int*)d_in[2];
    const int*   combining  = (const int*)d_in[3];
    const int*   seg_ids    = (const int*)d_in[4];
    const float* cat_emb    = (const float*)d_in[5];
    const float* case_emb   = (const float*)d_in[6];
    const float* comb_emb   = (const float*)d_in[7];
    const float* ln_g       = (const float*)d_in[8];
    const float* ln_b       = (const float*)d_in[9];
    const float* in_w       = (const float*)d_in[10];
    const float* in_b       = (const float*)d_in[11];
    const float* out_w      = (const float*)d_in[12];
    const float* out_b      = (const float*)d_in[13];
    float* out = (float*)d_out;

    char* ws = (char*)d_ws;
    float*          xn_f32   = (float*)ws;           ws += (size_t)S * D * sizeof(float);
    unsigned short* xn_bf    = (unsigned short*)ws;  ws += (size_t)S * D * 2;
    unsigned short* q_bf     = (unsigned short*)ws;  ws += (size_t)S * D * 2;
    unsigned short* k_bf     = (unsigned short*)ws;  ws += (size_t)S * D * 2;
    unsigned short* vT_bf    = (unsigned short*)ws;  ws += (size_t)D * S * 2;
    unsigned short* o_bf     = (unsigned short*)ws;  ws += (size_t)S * D * 2;
    unsigned short* in_w_bf  = (unsigned short*)ws;  ws += (size_t)3 * D * D * 2;
    unsigned short* out_w_bf = (unsigned short*)ws;

    {
        size_t total = (size_t)3 * D * D + (size_t)D * D;
        int blocks = (int)((total + 255) / 256);
        convert_w_kernel<<<blocks, 256, 0, stream>>>(in_w, out_w, in_w_bf, out_w_bf);
    }
    embed_ln_kernel<<<S, 256, 0, stream>>>(byte_repr, categories, casev, combining,
                                           cat_emb, case_emb, comb_emb, ln_g, ln_b,
                                           xn_f32, xn_bf);
    qkv_gemm_kernel<<<768, 256, 0, stream>>>(xn_bf, in_w_bf, in_b, q_bf, k_bf, vT_bf);
    attn_kernel<<<256, 256, 0, stream>>>(q_bf, k_bf, vT_bf, seg_ids, o_bf);
    out_gemm_kernel<<<256, 256, 0, stream>>>(o_bf, out_w_bf, out_b, xn_f32, out);
}
